// InsertionEdgeHead_13537736917286
// MI455X (gfx1250) — matrix operand order repaired and checked by reading, not compiled, not run on hardware
//
#include <hip/hip_runtime.h>
#include <hip/hip_bf16.h>
#include <math.h>

// ---------------------------------------------------------------------------
// InsertionEdgeHead on MI455X (gfx1250, wave32, WMMA).
//
// 64-edge M-tile per workgroup (256 threads = 8 waves). Each wave computes
// 4 row-tiles x 2 col-tiles (GEMM1) so every B fragment (W1t, L2-resident)
// feeds 4 v_wmma_f32_16x16x32_bf16 ops: W1t L2 traffic drops 4x vs a 16-edge
// tile (~9 GB -> ~2.3 GB), keeping the kernel bound by the unavoidable
// ~1 GB of random h-row gathers (~44 us @ 23.3 TB/s HBM).
// bf16 WMMA inputs, f32 accumulation; LN/SiLU/bias in f32.
// ---------------------------------------------------------------------------

typedef __attribute__((ext_vector_type(16))) __bf16        v16bf;
typedef __attribute__((ext_vector_type(8)))  float         v8f;
typedef __attribute__((ext_vector_type(4)))  unsigned int  v4u;

#define HID    256
#define H2     128
#define K1     551
#define K1PAD  576        // 18 * 32
#define E_TILE 64
#define NTHREADS 256
#define NEDGE_TYPES 5
#define LN_EPS 1e-5f

union mreg { v16bf v; v4u u[2]; };

// ---------------------------------------------------------------------------
// Prep: W[K][N] f32 (row-major)  ->  Wt[N][Kpad] bf16 (zero pad K)
// ---------------------------------------------------------------------------
__global__ void k_transpose_pad_bf16(const float* __restrict__ W,
                                     __bf16* __restrict__ Wt,
                                     int K, int N, int Kpad) {
  int idx = blockIdx.x * blockDim.x + threadIdx.x;
  if (idx >= N * Kpad) return;
  int n = idx / Kpad;
  int k = idx - n * Kpad;
  Wt[idx] = (k < K) ? (__bf16)W[k * N + n] : (__bf16)0.0f;
}

// ---------------------------------------------------------------------------
// Main fused kernel: 64 edges per workgroup, 256 threads (8 waves).
// ---------------------------------------------------------------------------
__global__ __launch_bounds__(NTHREADS) void edge_mlp_kernel(
    const float* __restrict__ h, const float* __restrict__ x,
    const int*   __restrict__ spawn, const int* __restrict__ exist,
    const float* __restrict__ ins_x, const int* __restrict__ ins_a,
    const int*   __restrict__ ins_c,
    const float* __restrict__ b1, const float* __restrict__ g1,
    const float* __restrict__ be1,
    const float* __restrict__ b2, const float* __restrict__ g2,
    const float* __restrict__ be2,
    const float* __restrict__ W3, const float* __restrict__ b3,
    const float* __restrict__ rbf_mu, const float* __restrict__ rbf_gamma,
    const __bf16* __restrict__ W1t,   // [256][576] bf16
    const __bf16* __restrict__ W2t,   // [128][256] bf16
    float* __restrict__ out, int E)
{
  // Region A (72 KB): phase1 = feats bf16 [64][576];
  //                   phase2 = A2 bf16 [64][256] (32 KB) + z2 f32 [64][128] (32 KB)
  __shared__ __align__(16) char  smemA[E_TILE * K1PAD * 2];
  __shared__ __align__(16) float sZ1[E_TILE * HID];          // 64 KB
  __shared__ float sRed[2 * E_TILE * 4];
  __shared__ float sMu[E_TILE], sRs[E_TILE], sD[E_TILE];

  __bf16* sA  = (__bf16*)smemA;                              // [64][576]
  __bf16* sA2 = (__bf16*)smemA;                              // [64][256] (alias)
  float*  sZ2 = (float*)(smemA + E_TILE * HID * 2);          // [64][128] @ 32 KB

  const int tid  = threadIdx.x;
  const int lane = tid & 31;
  const int wave = tid >> 5;    // 0..7
  const int row  = lane & 15;   // M (A) or N column (B/C)
  const int hi   = lane >> 4;   // half selector per WMMA VGPR layout
  const int e0   = blockIdx.x * E_TILE;

  // ---------------- build feature tile (bf16) in LDS ----------------------
  // gather h_spawn -> [0,256), h_existing -> [256,512)
  for (int i = tid; i < E_TILE * 128; i += NTHREADS) {   // 128 float4 per edge
    int e  = i >> 7;
    int j  = i & 127;
    int eg = min(e0 + e, E - 1);
    int node = (j < 64) ? spawn[eg] : exist[eg];
    float4 v = ((const float4*)(h + node * HID))[j & 63];
    __bf16* p = sA + e * K1PAD + ((j < 64) ? 0 : HID) + (j & 63) * 4;
    p[0] = (__bf16)v.x; p[1] = (__bf16)v.y; p[2] = (__bf16)v.z; p[3] = (__bf16)v.w;
  }
  // zero [512,576): rbf/one-hot/pad region
  for (int i = tid; i < E_TILE * 64; i += NTHREADS) {
    int e = i >> 6, k = i & 63;
    sA[e * K1PAD + 512 + k] = (__bf16)0.0f;
  }
  // per-edge distance
  if (tid < E_TILE) {
    int eg = min(e0 + tid, E - 1);
    int node = exist[eg];
    float dx = ins_x[eg * 3 + 0] - x[node * 3 + 0];
    float dy = ins_x[eg * 3 + 1] - x[node * 3 + 1];
    float dz = ins_x[eg * 3 + 2] - x[node * 3 + 2];
    sD[tid] = fmaxf(sqrtf(dx * dx + dy * dy + dz * dz), 1e-6f);
  }
  __syncthreads();
  // Gaussian RBF -> [512,528)
  {
    float gam = rbf_gamma[0];
    for (int i = tid; i < E_TILE * 16; i += NTHREADS) {
      int e = i >> 4, k = i & 15;
      float t = sD[e] - rbf_mu[k];
      sA[e * K1PAD + 512 + k] = (__bf16)__expf(-gam * t * t);
    }
  }
  // one-hots: atom -> [528,544), charge -> [544,551)
  if (tid < E_TILE) {
    int eg = min(e0 + tid, E - 1);
    sA[tid * K1PAD + 528 + ins_a[eg]] = (__bf16)1.0f;
    sA[tid * K1PAD + 544 + ins_c[eg]] = (__bf16)1.0f;
  }
  __syncthreads();

  // ---------------- GEMM1: feats[64x576] @ W1[576x256] --------------------
  // wave -> col-tiles {2w, 2w+1}; all 4 row-tiles. B fragment reused 4x.
  const v8f vzero = {0.f, 0.f, 0.f, 0.f, 0.f, 0.f, 0.f, 0.f};
  v8f acc1[4][2];
#pragma unroll
  for (int rt = 0; rt < 4; ++rt) { acc1[rt][0] = vzero; acc1[rt][1] = vzero; }

  for (int kk = 0; kk < K1PAD / 32; ++kk) {
    mreg a[4];                            // A 16x32 bf16 fragments (LDS)
#pragma unroll
    for (int rt = 0; rt < 4; ++rt) {
      const __bf16* ap = sA + (rt * 16 + row) * K1PAD + kk * 32 + hi * 8;
      a[rt].u[0] = *(const v4u*)(ap);     // K: base+0..7
      a[rt].u[1] = *(const v4u*)(ap + 16);// K: base+16..23
    }
#pragma unroll
    for (int c = 0; c < 2; ++c) {
      int colg = (wave * 2 + c) * 16 + row;
      mreg b;                             // B 32x16 bf16 fragment (global/L2)
      const __bf16* bp = W1t + colg * K1PAD + kk * 32 + hi * 8;
      b.u[0] = *(const v4u*)(bp);
      b.u[1] = *(const v4u*)(bp + 16);
#pragma unroll
      for (int rt = 0; rt < 4; ++rt)
        acc1[rt][c] = __builtin_amdgcn_wmma_f32_16x16x32_bf16(
            false, a[rt].v, false, b.v, (short)0, acc1[rt][c], false, false);
    }
  }
  // scatter C (+bias) to LDS: lane row = N col, VGPR r = M row (+8 if hi)
#pragma unroll
  for (int c = 0; c < 2; ++c) {
    int col = (wave * 2 + c) * 16 + row;
    float bias = b1[col];
#pragma unroll
    for (int rt = 0; rt < 4; ++rt)
#pragma unroll
      for (int r = 0; r < 8; ++r)
        sZ1[(rt * 16 + r + hi * 8) * HID + col] = acc1[rt][c][r] + bias;
  }
  __syncthreads();

  // ---------------- LN1 + SiLU -> bf16 A2 (aliases feats) -----------------
  {
    int rr = tid >> 2, seg = tid & 3;                 // 4 segs x 64 elems
    const float* zr = sZ1 + rr * HID + seg * 64;
    float s = 0.f, q = 0.f;
#pragma unroll
    for (int i = 0; i < 64; ++i) { float v = zr[i]; s += v; q += v * v; }
    sRed[rr * 4 + seg] = s; sRed[E_TILE * 4 + rr * 4 + seg] = q;
  }
  __syncthreads();
  if (tid < E_TILE) {
    float s = 0.f, q = 0.f;
    for (int j = 0; j < 4; ++j) { s += sRed[tid * 4 + j]; q += sRed[E_TILE * 4 + tid * 4 + j]; }
    float mu  = s * (1.0f / HID);
    float var = q * (1.0f / HID) - mu * mu;
    sMu[tid] = mu; sRs[tid] = rsqrtf(var + LN_EPS);
  }
  __syncthreads();
  for (int i = tid; i < E_TILE * HID; i += NTHREADS) {
    int m = i >> 8, col = i & 255;
    float v  = (sZ1[i] - sMu[m]) * sRs[m] * g1[col] + be1[col];
    float sv = v / (1.0f + __expf(-v));
    sA2[i] = (__bf16)sv;      // overwrites dead feats region
  }
  __syncthreads();

  // ---------------- GEMM2: z1[64x256] @ W2[256x128] -----------------------
  // wave -> col-tile w; all 4 row-tiles. B fragment reused 4x.
  v8f acc2[4];
#pragma unroll
  for (int rt = 0; rt < 4; ++rt) acc2[rt] = vzero;

  for (int kk = 0; kk < HID / 32; ++kk) {
    mreg a[4];
#pragma unroll
    for (int rt = 0; rt < 4; ++rt) {
      const __bf16* ap = sA2 + (rt * 16 + row) * HID + kk * 32 + hi * 8;
      a[rt].u[0] = *(const v4u*)(ap);
      a[rt].u[1] = *(const v4u*)(ap + 16);
    }
    int colg = wave * 16 + row;
    mreg b;
    const __bf16* bp = W2t + colg * HID + kk * 32 + hi * 8;
    b.u[0] = *(const v4u*)(bp);
    b.u[1] = *(const v4u*)(bp + 16);
#pragma unroll
    for (int rt = 0; rt < 4; ++rt)
      acc2[rt] = __builtin_amdgcn_wmma_f32_16x16x32_bf16(
          false, a[rt].v, false, b.v, (short)0, acc2[rt], false, false);
  }
  {
    int col = wave * 16 + row;
    float bias = b2[col];
#pragma unroll
    for (int rt = 0; rt < 4; ++rt)
#pragma unroll
      for (int r = 0; r < 8; ++r)
        sZ2[(rt * 16 + r + hi * 8) * H2 + col] = acc2[rt][r] + bias;
  }
  __syncthreads();

  // ---------------- LN2 + SiLU (in place, f32) ----------------------------
  {
    int rr = tid >> 2, seg = tid & 3;                 // 4 segs x 32 elems
    const float* zr = sZ2 + rr * H2 + seg * 32;
    float s = 0.f, q = 0.f;
#pragma unroll
    for (int i = 0; i < 32; ++i) { float v = zr[i]; s += v; q += v * v; }
    sRed[rr * 4 + seg] = s; sRed[E_TILE * 4 + rr * 4 + seg] = q;
  }
  __syncthreads();
  if (tid < E_TILE) {
    float s = 0.f, q = 0.f;
    for (int j = 0; j < 4; ++j) { s += sRed[tid * 4 + j]; q += sRed[E_TILE * 4 + tid * 4 + j]; }
    float mu  = s * (1.0f / H2);
    float var = q * (1.0f / H2) - mu * mu;
    sMu[tid] = mu; sRs[tid] = rsqrtf(var + LN_EPS);
  }
  __syncthreads();
  for (int i = tid; i < E_TILE * H2; i += NTHREADS) {
    int m = i >> 7, col = i & 127;
    float v = (sZ2[i] - sMu[m]) * sRs[m] * g2[col] + be2[col];
    sZ2[i] = v / (1.0f + __expf(-v));
  }
  __syncthreads();

  // ---------------- GEMM3: z2[64x128] @ W3[128x5] + b3 --------------------
  for (int i = tid; i < E_TILE * NEDGE_TYPES; i += NTHREADS) {
    int e = i / NEDGE_TYPES, cc = i - e * NEDGE_TYPES;
    int eg = e0 + e;
    if (eg < E) {
      float s = b3[cc];
      const float* zr = sZ2 + e * H2;
#pragma unroll 8
      for (int k = 0; k < H2; ++k) s += zr[k] * W3[k * NEDGE_TYPES + cc];
      out[eg * NEDGE_TYPES + cc] = s;
    }
  }
}

// ---------------------------------------------------------------------------
extern "C" void kernel_launch(void* const* d_in, const int* in_sizes, int n_in,
                              void* d_out, int out_size, void* d_ws, size_t ws_size,
                              hipStream_t stream) {
  const float* h        = (const float*)d_in[0];
  const float* x        = (const float*)d_in[1];
  const int*   spawn    = (const int*)  d_in[2];
  const int*   exist    = (const int*)  d_in[3];
  const float* ins_x    = (const float*)d_in[4];
  const int*   ins_a    = (const int*)  d_in[5];
  const int*   ins_c    = (const int*)  d_in[6];
  const float* W1       = (const float*)d_in[7];
  const float* b1       = (const float*)d_in[8];
  const float* g1       = (const float*)d_in[9];
  const float* be1      = (const float*)d_in[10];
  const float* W2       = (const float*)d_in[11];
  const float* b2       = (const float*)d_in[12];
  const float* g2       = (const float*)d_in[13];
  const float* be2      = (const float*)d_in[14];
  const float* W3       = (const float*)d_in[15];
  const float* b3       = (const float*)d_in[16];
  const float* rbf_mu   = (const float*)d_in[17];
  const float* rbf_gam  = (const float*)d_in[18];
  float*       out      = (float*)d_out;

  const int E = in_sizes[2];

  // workspace layout: W1t bf16 [256][576], W2t bf16 [128][256]
  __bf16* W1t = (__bf16*)d_ws;
  __bf16* W2t = (__bf16*)((char*)d_ws + (size_t)HID * K1PAD * sizeof(__bf16));

  {
    int t1 = HID * K1PAD;            // 147456
    k_transpose_pad_bf16<<<(t1 + 255) / 256, 256, 0, stream>>>(W1, W1t, K1, HID, K1PAD);
    int t2 = H2 * HID;               // 32768
    k_transpose_pad_bf16<<<(t2 + 255) / 256, 256, 0, stream>>>(W2, W2t, HID, H2, HID);
  }

  int nblocks = (E + E_TILE - 1) / E_TILE;   // 7813 for E=500000
  edge_mlp_kernel<<<nblocks, NTHREADS, 0, stream>>>(
      h, x, spawn, exist, ins_x, ins_a, ins_c,
      b1, g1, be1, b2, g2, be2, W3, b3, rbf_mu, rbf_gam,
      W1t, W2t, out, E);
}
